// GraphReconstructor_37769942401723
// MI455X (gfx1250) — compile-verified, matrix operand order checked by reading, not run on hardware
//
#include <hip/hip_runtime.h>
#include <stdint.h>

#define Bb 32
#define Pp 512
#define Ss 1024
#define Hh 1024
#define Rr 256
#define NCHUNK (Hh / 32)

typedef __attribute__((ext_vector_type(16))) _Float16 v16h;
typedef __attribute__((ext_vector_type(8)))  _Float16 v8h;
typedef __attribute__((ext_vector_type(8)))  float    v8f;
typedef __attribute__((ext_vector_type(4)))  float    v4f;

// Pointer types matching the async-to-LDS builtin signature (probed round 2):
// param0 = v4i addrspace(1)*, param1 = v4i addrspace(3)*.
typedef int v4i __attribute__((__vector_size__(4 * sizeof(int))));
typedef __attribute__((address_space(1))) v4i* gbl_v4i_p;
typedef __attribute__((address_space(3))) v4i* lds_v4i_p;

#if __has_builtin(__builtin_amdgcn_global_load_async_to_lds_b128) && \
    __has_builtin(__builtin_amdgcn_s_wait_asynccnt)
#define USE_ASYNC_LDS 1
#else
#define USE_ASYNC_LDS 0
#endif

// ---------------------------------------------------------------------------
// Prep: convert W [R, 3H] f32 -> f16, permuted chunk-major:
//   Wp[c][r][s*32 + j] = W[r][s*H + c*32 + j]
// so each main-kernel chunk reads a fully contiguous 48KB block.
// ---------------------------------------------------------------------------
__global__ __launch_bounds__(256)
void wperm_kernel(const float* __restrict__ W, _Float16* __restrict__ Wp) {
  int idx = blockIdx.x * 256 + threadIdx.x;
  if (idx >= Rr * 3 * Hh) return;
  int j  = idx % 96;
  int r  = (idx / 96) % Rr;
  int c  = idx / (96 * Rr);
  int s  = j >> 5;
  int jj = j & 31;
  Wp[idx] = (_Float16)W[(size_t)r * (3 * Hh) + (size_t)s * Hh + c * 32 + jj];
}

// ---------------------------------------------------------------------------
// Fused gather + GEMM:  out[b,p,r] = sum_h head*W[r,h] + tail*W[r,H+h]
//                                    + head*tail*W[r,2H+h]  + bias[r]
// Double-buffered software pipeline, one barrier per chunk:
//   iter c: stage A(c+1) (ds) | issue async W(c+1) | gather regs(c+2) |
//           24 WMMAs on buffers[c&1] | s_wait_asynccnt 0 | barrier
// ---------------------------------------------------------------------------
__global__ __launch_bounds__(256)
void fused_pair_gemm(const int*      __restrict__ pairs,
                     const float*    __restrict__ hs,
                     const _Float16* __restrict__ Wp,
                     const float*    __restrict__ bias,
                     float*          __restrict__ out) {
  __shared__ _Float16 Ah[2][64][40];    // head tiles (80B row stride)
  __shared__ _Float16 At[2][64][40];    // tail tiles
  __shared__ _Float16 Wt[2][256][104];  // W chunks: 256 rows x 96 cols (+pad)
  __shared__ int      idx4[64][4];

  const int tid   = threadIdx.x;
  const int lane  = tid & 31;
  const int wave  = tid >> 5;
  const int b     = blockIdx.x >> 3;
  const int pbase = (blockIdx.x & 7) * 64;

  const int mSub = wave & 3;   // 16-row subtile within the 64-row tile
  const int nGrp = wave >> 2;  // low/high 128 columns

  if (tid < 64) {
    ((int4*)idx4)[tid] = ((const int4*)pairs)[(size_t)b * Pp + pbase + tid];
  }
  __syncthreads();

  // Per-thread gather pointers: thread = (row 0..63, colgroup 0..3 of 8 floats)
  const int arow = tid >> 2;
  const int cg   = tid & 3;
  const float* hb  = hs + (size_t)b * Ss * Hh;
  const float* p00 = hb + (size_t)idx4[arow][0] * Hh + cg * 8;
  const float* p01 = hb + (size_t)idx4[arow][1] * Hh + cg * 8;
  const float* p10 = hb + (size_t)idx4[arow][2] * Hh + cg * 8;
  const float* p11 = hb + (size_t)idx4[arow][3] * Hh + cg * 8;

  const int lo  = lane & 15;   // N (or M) position within 16
  const int sel = lane >> 4;   // K-half select per WMMA 16-bit layout

  // W-staging thread mapping: 12 x 16B units per thread
  const int uR[1] = {0}; (void)uR;

  v8f acc[8] = {};
  v4f x0, x1, y0, y1, z0, z1, w0, w1;

  // ---- helpers as lambdas -------------------------------------------------
  auto gatherRegs = [&](int c) {
    const int hh_ = c * 32;
    x0 = *(const v4f*)(p00 + hh_);  x1 = *(const v4f*)(p00 + hh_ + 4);
    y0 = *(const v4f*)(p01 + hh_);  y1 = *(const v4f*)(p01 + hh_ + 4);
    z0 = *(const v4f*)(p10 + hh_);  z1 = *(const v4f*)(p10 + hh_ + 4);
    w0 = *(const v4f*)(p11 + hh_);  w1 = *(const v4f*)(p11 + hh_ + 4);
  };
  auto stageA = [&](int buf) {
    v8h hh, tt;
#pragma unroll
    for (int i = 0; i < 4; ++i) {
      hh[i]     = (_Float16)(0.5f * (x0[i] + y0[i]));
      hh[i + 4] = (_Float16)(0.5f * (x1[i] + y1[i]));
      tt[i]     = (_Float16)(0.5f * (z0[i] + w0[i]));
      tt[i + 4] = (_Float16)(0.5f * (z1[i] + w1[i]));
    }
    *(v8h*)&Ah[buf][arow][cg * 8] = hh;
    *(v8h*)&At[buf][arow][cg * 8] = tt;
  };
  auto issueW = [&](int c, int buf) {
    const _Float16* wsrc = Wp + (size_t)c * (Rr * 96);
#pragma unroll
    for (int i = 0; i < 12; ++i) {
      int u = i * 256 + tid;           // 16-byte unit id, 0..3071
      int r = u / 12, q = u % 12;
#if USE_ASYNC_LDS
      __builtin_amdgcn_global_load_async_to_lds_b128(
          (gbl_v4i_p)(wsrc + (size_t)u * 8),
          (lds_v4i_p)&Wt[buf][r][q * 8],
          0, 0);
#else
      *(v8h*)&Wt[buf][r][q * 8] = *(const v8h*)(wsrc + (size_t)u * 8);
#endif
    }
  };

  // ---- pipeline prologue: prime buffer 0 ---------------------------------
  gatherRegs(0);
  stageA(0);
  issueW(0, 0);
  gatherRegs(1);
#if USE_ASYNC_LDS
  __builtin_amdgcn_s_wait_asynccnt(0);
#endif
  __syncthreads();   // A(0), W(0) visible to all waves

  for (int c = 0; c < NCHUNK; ++c) {
    const int cb = c & 1;

    // ---- stage next chunk into the other buffers (overlaps with compute) ----
    if (c + 1 < NCHUNK) {
      stageA((c + 1) & 1);
      issueW(c + 1, (c + 1) & 1);
    }
    if (c + 2 < NCHUNK) gatherRegs(c + 2);

    // ---- A fragments (16-bit layout: lane<16 -> K0..7,16..23; lane>=16 -> K8..15,24..31)
    const int mrow = mSub * 16 + lo;
    union Frag { v16h v; v8h h[2]; };
    Frag fh, ft;
    fh.h[0] = *(const v8h*)&Ah[cb][mrow][sel * 8];
    fh.h[1] = *(const v8h*)&Ah[cb][mrow][sel * 8 + 16];
    ft.h[0] = *(const v8h*)&At[cb][mrow][sel * 8];
    ft.h[1] = *(const v8h*)&At[cb][mrow][sel * 8 + 16];
    v16h fp = fh.v * ft.v;   // head*tail fragment (identical element mapping)

#pragma unroll
    for (int t = 0; t < 8; ++t) {
      const int nrow = (nGrp * 8 + t) * 16 + lo;
      Frag wa, wb, wc;
      wa.h[0] = *(const v8h*)&Wt[cb][nrow][ 0 + sel * 8];
      wa.h[1] = *(const v8h*)&Wt[cb][nrow][16 + sel * 8];
      wb.h[0] = *(const v8h*)&Wt[cb][nrow][32 + sel * 8];
      wb.h[1] = *(const v8h*)&Wt[cb][nrow][48 + sel * 8];
      wc.h[0] = *(const v8h*)&Wt[cb][nrow][64 + sel * 8];
      wc.h[1] = *(const v8h*)&Wt[cb][nrow][80 + sel * 8];
      acc[t] = __builtin_amdgcn_wmma_f32_16x16x32_f16(false, fh.v, false, wa.v,
                                                      (short)0, acc[t], false, false);
      acc[t] = __builtin_amdgcn_wmma_f32_16x16x32_f16(false, ft.v, false, wb.v,
                                                      (short)0, acc[t], false, false);
      acc[t] = __builtin_amdgcn_wmma_f32_16x16x32_f16(false, fp,   false, wc.v,
                                                      (short)0, acc[t], false, false);
    }

    // ---- close the pipeline stage: next buffers ready for everyone ----
#if USE_ASYNC_LDS
    __builtin_amdgcn_s_wait_asynccnt(0);   // W(c+1) landed (overlapped compute)
#endif
    __syncthreads();                        // + A(c+1) ds stores visible
  }

  // ---- epilogue: C/D layout -> VGPR v holds M = v + 8*sel, N = lo ----
  const int rowTop = pbase + mSub * 16 + sel * 8;
#pragma unroll
  for (int t = 0; t < 8; ++t) {
    const int nb  = (nGrp * 8 + t) * 16;
    const float bv = bias[nb + lo];
    float* o = out + ((size_t)(b * Pp + rowTop)) * Rr + nb + lo;
#pragma unroll
    for (int v = 0; v < 8; ++v) o[(size_t)v * Rr] = acc[t][v] + bv;
  }
}

// ---------------------------------------------------------------------------
extern "C" void kernel_launch(void* const* d_in, const int* in_sizes, int n_in,
                              void* d_out, int out_size, void* d_ws, size_t ws_size,
                              hipStream_t stream) {
  (void)in_sizes; (void)n_in; (void)out_size; (void)ws_size;
  const int*   pairs = (const int*)d_in[0];
  const float* hsb   = (const float*)d_in[1];
  const float* W     = (const float*)d_in[2];
  const float* bias  = (const float*)d_in[3];
  float*       out   = (float*)d_out;
  _Float16*    Wp    = (_Float16*)d_ws;   // 32*256*96 f16 = 1.5 MB scratch

  wperm_kernel<<<(Rr * 3 * Hh + 255) / 256, 256, 0, stream>>>(W, Wp);
  fused_pair_gemm<<<Bb * (Pp / 64), 256, 0, stream>>>(pairs, hsb, Wp, bias, out);
}
